// Attention_41180146434858
// MI455X (gfx1250) — matrix operand order repaired and checked by reading, not compiled, not run on hardware
//
#include <hip/hip_runtime.h>
#include <cmath>

// ---------------------------------------------------------------------------
// Attention forward for MI455X (gfx1250): wave32 + WMMA bf16 + TDM tile DMA.
//   q = x@wq^T ; k = x@wk^T ; v = x@wv^T           (WMMA GEMMs, bf16 in / f32 acc)
//   rope(q), rope(k)
//   per head: scores = q@k^T * 1/sqrt(128) + causal ; softmax ; out = probs@v
//   result = attn @ wo^T
// GEMM main loop: tensor_load_to_lds (Tensor Data Mover, TENSORcnt) feeds
// double-buffered LDS; ds_load fragments + v_wmma_f32_16x16x32_bf16 compute.
// ---------------------------------------------------------------------------

typedef __attribute__((ext_vector_type(16))) __bf16 v16bf;
typedef __attribute__((ext_vector_type(8)))  float  v8f;
typedef unsigned int u32x4 __attribute__((ext_vector_type(4)));
typedef unsigned int u32x8 __attribute__((ext_vector_type(8)));

#define S_LEN    2048
#define D_DIM    2048
#define N_HEADS  16
#define D_HEAD   128
#define ROT_HALF 64

#define BM 128
#define BN 128
#define BK 32
#define LDSP 40   // padded LDS row stride (bf16): 64B row + 16B pad = 80B = 5*16B

union Frag16 { v16bf bf; uint4 q[2]; };
union FragC  { v8f f; float e[8]; };

__device__ __forceinline__ unsigned short f32_to_bf16(float f) {
  unsigned int u = __float_as_uint(f);
  u += 0x7FFFu + ((u >> 16) & 1u);          // round to nearest even
  return (unsigned short)(u >> 16);
}
__device__ __forceinline__ float bf16_to_f32(unsigned short h) {
  return __uint_as_float(((unsigned int)h) << 16);
}

// 32-bit LDS (addrspace 3) byte offset of a __shared__ object
__device__ __forceinline__ unsigned lds_addr(const void* p) {
  return (unsigned)(unsigned long long)(__attribute__((address_space(3))) const void*)p;
}

// --- Tensor Data Mover: DMA one BMxBK bf16 tile (row stride ld elems) from
// global memory into LDS with 16B padding after each 64B row (-> LDSP stride).
// D# groups built in SGPRs; 2D tile so groups 2/3 are omitted (NULL).
__device__ __forceinline__ void tdm_load_tile(unsigned lds_byte_addr,
                                              const unsigned short* gptr,
                                              unsigned ld_elems) {
  unsigned long long ga = (unsigned long long)gptr;
  u32x4 g0;
  g0[0] = 1u;                                   // count=1, user descriptor
  g0[1] = lds_byte_addr;                        // lds_addr
  g0[2] = (unsigned)ga;                         // global_addr[31:0]
  g0[3] = (unsigned)((ga >> 32) & 0x01FFFFFFu)  // global_addr[56:32]
          | (2u << 30);                         // type=2 ("image")
  u32x8 g1;
  g1[0] = (1u << 16)      // data_size = 1 -> 2 bytes
        | (1u << 20)      // pad_enable
        | (3u << 22)      // pad_interval = 3 -> every 16 DWORDs (64B row)
        | (3u << 25);     // pad_amount   = 3 -> 4 DWORDs (16B)
  g1[1] = ((unsigned)BK) << 16;        // tensor_dim0[15:0] = 32
  g1[2] = (((unsigned)BM) << 16);      // tensor_dim0 hi = 0 | tensor_dim1 lo = 128
  g1[3] = (((unsigned)BK) << 16);      // tensor_dim1 hi = 0 | tile_dim0 = 32
  g1[4] = (unsigned)BM;                // tile_dim1 = 128 | tile_dim2 = 0
  g1[5] = ld_elems;                    // tensor_dim0_stride[31:0]
  g1[6] = 0u;                          // stride hi | tensor_dim1_stride lo (unused)
  g1[7] = 0u;
  asm volatile("tensor_load_to_lds %0, %1" :: "s"(g0), "s"(g1) : "memory");
}

// ---------------- f32 -> bf16 bulk convert (vectorized x4) -----------------
__global__ __launch_bounds__(256)
void cvt_f32_to_bf16_x4(const float* __restrict__ in,
                        unsigned short* __restrict__ out, int n4) {
  int i = blockIdx.x * blockDim.x + threadIdx.x;
  if (i >= n4) return;
  float4 v = ((const float4*)in)[i];
  unsigned int lo = (unsigned int)f32_to_bf16(v.x) |
                    ((unsigned int)f32_to_bf16(v.y) << 16);
  unsigned int hi = (unsigned int)f32_to_bf16(v.z) |
                    ((unsigned int)f32_to_bf16(v.w) << 16);
  ((uint2*)out)[i] = make_uint2(lo, hi);
}

// ---------------- WMMA GEMM: C[M,N] = A[M,K] * B[N,K]^T --------------------
// A, B bf16 row-major with K contiguous. 256 threads = 8 waves (wave32),
// block tile 128x128, wave tile 32x64 (2x4 WMMA 16x16x32 tiles), BK=32,
// double-buffered LDS fed by the Tensor Data Mover (wave 0 issues, all wait).
// epilogue: 0 = bf16 row-major C[m*ldc+n]
//           1 = bf16 transposed C[n*ldc+m]  (vectorized 8xbf16 store)
//           2 = f32 C[m*ldc+n] = acc*scale + causal_mask(m,n)
//           3 = f32 C[m*ldc+n] = acc
// grid.z = heads; per-head element offsets saH/sbH/scH.
__global__ __launch_bounds__(256)
void gemm_bt_bf16(const unsigned short* __restrict__ A, int lda, long long saH,
                  const unsigned short* __restrict__ B, int ldb, long long sbH,
                  void* __restrict__ Cout, int ldc, long long scH,
                  int M, int N, int K, int epilogue, float scale) {
  __shared__ __align__(16) unsigned short As[2][BM * LDSP];
  __shared__ __align__(16) unsigned short Bs[2][BN * LDSP];

  const int h = blockIdx.z;
  const unsigned short* Ah = A + saH * h;
  const unsigned short* Bh = B + sbH * h;

  const int tid   = threadIdx.x;
  const int wave  = tid >> 5;
  const int lane  = tid & 31;
  const int waveM = wave & 3;   // 0..3 -> 4 wave rows
  const int waveN = wave >> 2;  // 0..1 -> 2 wave cols
  const int l16   = lane & 15;
  const int lhalf = lane >> 4;

  const int m0 = blockIdx.y * BM;
  const int n0 = blockIdx.x * BN;
  const unsigned short* Atile = Ah + (long long)m0 * lda;  // K-major row start
  const unsigned short* Btile = Bh + (long long)n0 * ldb;

  FragC acc[2][4];
#pragma unroll
  for (int mi = 0; mi < 2; mi++)
#pragma unroll
    for (int ni = 0; ni < 4; ni++)
#pragma unroll
      for (int v = 0; v < 8; v++) acc[mi][ni].e[v] = 0.0f;

  // prologue: TDM-fill stage 0 with the k0=0 tiles (wave 0 only: TDM ignores
  // EXEC and is per-wave -- issuing from every wave would duplicate the DMA)
  if (wave == 0) {
    tdm_load_tile(lds_addr(&As[0][0]), Atile, (unsigned)lda);
    tdm_load_tile(lds_addr(&Bs[0][0]), Btile, (unsigned)ldb);
  }
  __builtin_amdgcn_s_wait_tensorcnt(0);   // no-op for non-issuing waves
  __syncthreads();

  int stage = 0;
  for (int k0 = 0; k0 < K; k0 += BK) {
    // prefetch next tile into the other buffer (overlaps with WMMA below)
    if (k0 + BK < K && wave == 0) {
      const int ns = stage ^ 1;
      const int kn = k0 + BK;
      tdm_load_tile(lds_addr(&As[ns][0]), Atile + kn, (unsigned)lda);
      tdm_load_tile(lds_addr(&Bs[ns][0]), Btile + kn, (unsigned)ldb);
    }

    // ---- load fragments per CDNA5 16-bit WMMA layouts ----
    // A (16x32): lane<16 holds K {0..7,16..23}; lane>=16 holds K {8..15,24..31}
    Frag16 af[2], bfr[4];
    const unsigned short* Asb = As[stage];
    const unsigned short* Bsb = Bs[stage];
#pragma unroll
    for (int mi = 0; mi < 2; mi++) {
      int r  = waveM * 32 + mi * 16 + l16;
      int kb = lhalf * 8;
      af[mi].q[0] = *(const uint4*)&Asb[r * LDSP + kb];
      af[mi].q[1] = *(const uint4*)&Asb[r * LDSP + 16 + kb];
    }
    // B (32x16): lane<16 -> col N=lane, K {0..7,16..23}; lane>=16 -> K {8..15,24..31}
#pragma unroll
    for (int ni = 0; ni < 4; ni++) {
      int c  = waveN * 64 + ni * 16 + l16;
      int kb = lhalf * 8;
      bfr[ni].q[0] = *(const uint4*)&Bsb[c * LDSP + kb];
      bfr[ni].q[1] = *(const uint4*)&Bsb[c * LDSP + 16 + kb];
    }

#pragma unroll
    for (int mi = 0; mi < 2; mi++)
#pragma unroll
      for (int ni = 0; ni < 4; ni++)
        acc[mi][ni].f = __builtin_amdgcn_wmma_f32_16x16x32_bf16(
            false, af[mi].bf, false, bfr[ni].bf, (short)0, acc[mi][ni].f,
            false, false);

    // next tile's DMA done (issuing wave), then block-wide visibility
    __builtin_amdgcn_s_wait_tensorcnt(0);
    __syncthreads();
    stage ^= 1;
  }

  // ---- epilogue; C 16x16 layout: lane<16 rows M=v, lane>=16 rows M=v+8 ----
  const int mW = m0 + waveM * 32;
  const int nW = n0 + waveN * 64;
#pragma unroll
  for (int mi = 0; mi < 2; mi++) {
#pragma unroll
    for (int ni = 0; ni < 4; ni++) {
      const int mb = mW + mi * 16 + lhalf * 8;
      const int n  = nW + ni * 16 + l16;
      const FragC& c = acc[mi][ni];
      if (epilogue == 0) {
        unsigned short* C = (unsigned short*)Cout + scH * h;
#pragma unroll
        for (int v = 0; v < 8; v++)
          C[(long long)(mb + v) * ldc + n] = f32_to_bf16(c.e[v]);
      } else if (epilogue == 1) {
        unsigned short* C = (unsigned short*)Cout + scH * h;
        unsigned int w0 = (unsigned int)f32_to_bf16(c.e[0]) |
                          ((unsigned int)f32_to_bf16(c.e[1]) << 16);
        unsigned int w1 = (unsigned int)f32_to_bf16(c.e[2]) |
                          ((unsigned int)f32_to_bf16(c.e[3]) << 16);
        unsigned int w2 = (unsigned int)f32_to_bf16(c.e[4]) |
                          ((unsigned int)f32_to_bf16(c.e[5]) << 16);
        unsigned int w3 = (unsigned int)f32_to_bf16(c.e[6]) |
                          ((unsigned int)f32_to_bf16(c.e[7]) << 16);
        *(uint4*)&C[(long long)n * ldc + mb] = make_uint4(w0, w1, w2, w3);
      } else if (epilogue == 2) {
        float* C = (float*)Cout + scH * h;
#pragma unroll
        for (int v = 0; v < 8; v++) {
          int m = mb + v;
          C[(long long)m * ldc + n] =
              c.e[v] * scale + ((n > m) ? -1.0e9f : 0.0f);
        }
      } else {
        float* C = (float*)Cout + scH * h;
#pragma unroll
        for (int v = 0; v < 8; v++)
          C[(long long)(mb + v) * ldc + n] = c.e[v];
      }
    }
  }
}

// ---------------- RoPE (in place on bf16 [S, D], interleaved pairs) --------
__global__ __launch_bounds__(256)
void rope_kernel(unsigned short* __restrict__ t,
                 const float* __restrict__ cosf,
                 const float* __restrict__ sinf) {
  int idx = blockIdx.x * blockDim.x + threadIdx.x;  // S*H*64 threads
  int i  = idx & (ROT_HALF - 1);
  int sh = idx >> 6;
  int hh = sh & (N_HEADS - 1);
  int s  = sh >> 4;
  float c  = cosf[s * ROT_HALF + i];
  float sn = sinf[s * ROT_HALF + i];
  long long base = (long long)s * D_DIM + hh * D_HEAD + 2 * i;
  float e = bf16_to_f32(t[base]);
  float o = bf16_to_f32(t[base + 1]);
  t[base]     = f32_to_bf16(e * c - o * sn);
  t[base + 1] = f32_to_bf16(e * sn + o * c);
}

// ---------------- row softmax: f32 scores row -> bf16 probs row ------------
__global__ __launch_bounds__(256)
void softmax_kernel(const float* __restrict__ scores,
                    unsigned short* __restrict__ probs) {
  __shared__ float red[256];
  const int row = blockIdx.x;
  const int tid = threadIdx.x;
  const float* r = scores + (long long)row * S_LEN;
  float mx = -3.0e38f;
  for (int j = tid; j < S_LEN; j += 256) mx = fmaxf(mx, r[j]);
  red[tid] = mx;
  __syncthreads();
  for (int off = 128; off > 0; off >>= 1) {
    if (tid < off) red[tid] = fmaxf(red[tid], red[tid + off]);
    __syncthreads();
  }
  const float m = red[0];
  __syncthreads();
  float sum = 0.0f;
  for (int j = tid; j < S_LEN; j += 256) sum += __expf(r[j] - m);
  red[tid] = sum;
  __syncthreads();
  for (int off = 128; off > 0; off >>= 1) {
    if (tid < off) red[tid] += red[tid + off];
    __syncthreads();
  }
  const float inv = 1.0f / red[0];
  unsigned short* pr = probs + (long long)row * S_LEN;
  for (int j = tid; j < S_LEN; j += 256)
    pr[j] = f32_to_bf16(__expf(r[j] - m) * inv);
}

// ---------------------------------------------------------------------------
extern "C" void kernel_launch(void* const* d_in, const int* in_sizes, int n_in,
                              void* d_out, int out_size, void* d_ws,
                              size_t ws_size, hipStream_t stream) {
  const float* x  = (const float*)d_in[0];
  const float* fc = (const float*)d_in[1];
  const float* fs = (const float*)d_in[2];
  // d_in[3] mask: causal mask reproduced inline in the scores epilogue
  const float* wq = (const float*)d_in[4];
  const float* wk = (const float*)d_in[5];
  const float* wv = (const float*)d_in[6];
  const float* wo = (const float*)d_in[7];
  // d_in[8] start_pos == 0, unused
  float* out = (float*)d_out;

  const long long SD = (long long)S_LEN * D_DIM;  // 4M elems
  // workspace layout (bf16 buffers are unsigned short):
  unsigned short* xb    = (unsigned short*)d_ws;   //  8 MB
  unsigned short* wqb   = xb  + SD;                //  8 MB
  unsigned short* wkb   = wqb + SD;
  unsigned short* wvb   = wkb + SD;
  unsigned short* wob   = wvb + SD;
  unsigned short* qb    = wob + SD;                // q (bf16, [S,D])
  unsigned short* kb    = qb  + SD;                // k (bf16, [S,D])
  unsigned short* vT    = kb  + SD;                // v^T (bf16, [D,S])
  unsigned short* attnT = vT  + SD;                // attn^T (bf16, [D,S])
  float* scores         = (float*)(attnT + SD);    // 16 MB, reused per head
  unsigned short* probs = (unsigned short*)(scores + (long long)S_LEN * S_LEN);
  // probs: 16 heads * S*S bf16 = 128 MB; total ws use ~216 MB

  const int n4 = (int)(SD / 4);
  dim3 cgrd((n4 + 255) / 256), blk(256);
  cvt_f32_to_bf16_x4<<<cgrd, blk, 0, stream>>>(x,  xb,  n4);
  cvt_f32_to_bf16_x4<<<cgrd, blk, 0, stream>>>(wq, wqb, n4);
  cvt_f32_to_bf16_x4<<<cgrd, blk, 0, stream>>>(wk, wkb, n4);
  cvt_f32_to_bf16_x4<<<cgrd, blk, 0, stream>>>(wv, wvb, n4);
  cvt_f32_to_bf16_x4<<<cgrd, blk, 0, stream>>>(wo, wob, n4);

  dim3 gproj(D_DIM / BN, S_LEN / BM, 1);
  // q = x @ wq^T  (bf16 row-major out)
  gemm_bt_bf16<<<gproj, blk, 0, stream>>>(xb, D_DIM, 0, wqb, D_DIM, 0,
                                          qb, D_DIM, 0, S_LEN, D_DIM, D_DIM,
                                          0, 1.0f);
  // k = x @ wk^T
  gemm_bt_bf16<<<gproj, blk, 0, stream>>>(xb, D_DIM, 0, wkb, D_DIM, 0,
                                          kb, D_DIM, 0, S_LEN, D_DIM, D_DIM,
                                          0, 1.0f);
  // v = x @ wv^T, stored transposed -> vT[d, s]
  gemm_bt_bf16<<<gproj, blk, 0, stream>>>(xb, D_DIM, 0, wvb, D_DIM, 0,
                                          vT, S_LEN, 0, S_LEN, D_DIM, D_DIM,
                                          1, 1.0f);

  const int rthreads = S_LEN * N_HEADS * ROT_HALF;  // 2M
  rope_kernel<<<rthreads / 256, blk, 0, stream>>>(qb, fc, fs);
  rope_kernel<<<rthreads / 256, blk, 0, stream>>>(kb, fc, fs);

  const float scale = 1.0f / sqrtf((float)D_HEAD);
  for (int h = 0; h < N_HEADS; h++) {
    dim3 gs(S_LEN / BN, S_LEN / BM, 1);
    // scores = q[h] @ k[h]^T * scale + causal_mask   (f32)
    gemm_bt_bf16<<<gs, blk, 0, stream>>>(qb + h * D_HEAD, D_DIM, 0,
                                         kb + h * D_HEAD, D_DIM, 0,
                                         scores, S_LEN, 0,
                                         S_LEN, S_LEN, D_HEAD, 2, scale);
    softmax_kernel<<<S_LEN, blk, 0, stream>>>(
        scores, probs + (long long)h * S_LEN * S_LEN);
  }

  // out[h] = probs[h] @ v[h]; stored transposed per head -> attnT[(h*128+d), s]
  dim3 gpv(1, S_LEN / BM, N_HEADS);  // N = D_HEAD = one BN tile
  gemm_bt_bf16<<<gpv, blk, 0, stream>>>(
      probs, S_LEN, (long long)S_LEN * S_LEN,
      vT, S_LEN, (long long)D_HEAD * S_LEN,
      attnT, S_LEN, (long long)D_HEAD * S_LEN,
      S_LEN, D_HEAD, S_LEN, 1, 1.0f);

  // result = attnT(viewed [S,D] row-major, matching reference reshape) @ wo^T
  gemm_bt_bf16<<<gproj, blk, 0, stream>>>(attnT, D_DIM, 0, wob, D_DIM, 0,
                                          out, D_DIM, 0, S_LEN, D_DIM, D_DIM,
                                          3, 1.0f);
}
